// ConvAttention_88596585382404
// MI455X (gfx1250) — matrix operand order repaired and checked by reading, not compiled, not run on hardware
//
#include <hip/hip_runtime.h>

#define HEADS    4
#define DHEAD    32
#define NB       4        // batch
#define NC       256      // channels
#define NN       4096     // h*w tokens
#define INNER    128      // HEADS*DHEAD
#define OQKV     384      // 3*INNER
#define QK_SCALE 0.17677669529663687f   // 32^-0.5

typedef __attribute__((ext_vector_type(16))) __bf16 v16bf;
typedef __attribute__((ext_vector_type(8)))  __bf16 v8bf;
typedef __attribute__((ext_vector_type(8)))  float  v8f;

static __device__ __forceinline__ v16bf join16(v8bf lo, v8bf hi) {
  union { v16bf v; v8bf h[2]; } u;
  u.h[0] = lo; u.h[1] = hi;
  return u.v;
}

// A-operand fragment (M16 x K32 bf16) from row-major [row][k] storage.
// ISA 7.12.2 (16-bit A 16x32): lanes 0-15 row M=lane hold K={0..7,16..23};
// lanes 16-31 row M=lane-16 hold K={8..15,24..31}.  Two 16B loads per lane.
static __device__ __forceinline__ v16bf load_fragA(const __bf16* base, int pitch, int lane) {
  const int row = lane & 15, half = lane >> 4;
  const __bf16* p = base + (size_t)row * pitch + half * 8;
  return join16(*(const v8bf*)p, *(const v8bf*)(p + 16));
}

// B-operand fragment (K32 x N16 bf16) loaded from B^T storage [n][k], k contiguous.
// lanes 0-15: col N=lane hold K={0..7,16..23}; lanes 16-31: col N=lane-16 hold K={8..15,24..31}.
static __device__ __forceinline__ v16bf load_fragB(const __bf16* baseT, int pitch, int lane) {
  const int row = lane & 15, half = lane >> 4;
  const __bf16* p = baseT + (size_t)row * pitch + half * 8;
  return join16(*(const v8bf*)p, *(const v8bf*)(p + 16));
}

static __device__ __forceinline__ v8f wmma_bf16f32(v16bf a, v16bf b, v8f c) {
  return __builtin_amdgcn_wmma_f32_16x16x32_bf16(false, a, false, b, (short)0, c, false, false);
}

// ---------------------------------------------------------------- kernel 1
// Convert w_qkv (384x256) and w_out (256x128) f32 -> bf16.
__global__ void cvt_weights_k(const float* __restrict__ wq,
                              const float* __restrict__ wo,
                              unsigned short* __restrict__ wq_bf_u,
                              unsigned short* __restrict__ wo_bf_u) {
  __bf16* wq_bf = (__bf16*)wq_bf_u;
  __bf16* wo_bf = (__bf16*)wo_bf_u;
  int i = blockIdx.x * blockDim.x + threadIdx.x;
  if (i < OQKV * NC)   wq_bf[i] = (__bf16)wq[i];
  if (i < NC * INNER)  wo_bf[i] = (__bf16)wo[i];
}

// ---------------------------------------------------------------- kernel 2
// x[b][c][n] f32 -> xT[b][n][c] bf16, 32x32 tiles through LDS.
__global__ void transpose_x_k(const float* __restrict__ x,
                              unsigned short* __restrict__ xT_u) {
  __bf16* xT = (__bf16*)xT_u;
  __shared__ float tile[32][33];
  const int b = blockIdx.z, c0 = blockIdx.y * 32, n0 = blockIdx.x * 32;
  const float* xp = x + ((size_t)b * NC + c0) * NN + n0;
  #pragma unroll
  for (int r = threadIdx.y; r < 32; r += 8)
    tile[r][threadIdx.x] = xp[(size_t)r * NN + threadIdx.x];   // tile[c_local][n_local]
  __syncthreads();
  __bf16* op = xT + ((size_t)b * NN + n0) * NC + c0;
  #pragma unroll
  for (int r = threadIdx.y; r < 32; r += 8)
    op[(size_t)r * NC + threadIdx.x] = (__bf16)tile[threadIdx.x][r]; // xT[n][c]
}

// ---------------------------------------------------------------- kernel 3
// QKV projection: qkv[b][o][n] = sum_c w_qkv[o][c] * x[b][c][n].
// One wave per (b, 16-row o-tile, 16-col n-tile); K loop over c in steps of 32.
// Q (scaled) and K stored [b][h][n][d]; V stored [b][h][d][n].
__global__ void qkv_proj_k(const unsigned short* __restrict__ wq_bf_u,
                           const unsigned short* __restrict__ xT_u,
                           unsigned short* __restrict__ Qm_u,
                           unsigned short* __restrict__ Km_u,
                           unsigned short* __restrict__ Vm_u) {
  const __bf16* wq_bf = (const __bf16*)wq_bf_u;
  const __bf16* xT    = (const __bf16*)xT_u;
  __bf16* Qm = (__bf16*)Qm_u;  __bf16* Km = (__bf16*)Km_u;  __bf16* Vm = (__bf16*)Vm_u;

  const int lane = threadIdx.x & 31;
  const int wave = (blockIdx.x * blockDim.x + threadIdx.x) >> 5;
  const int nt = wave & 255;           // 256 n-tiles
  const int t2 = wave >> 8;
  const int ot = t2 % 24;              // 24 o-tiles
  const int b  = t2 / 24;
  const int o0 = ot * 16, n0 = nt * 16;

  v8f acc;
  #pragma unroll
  for (int r = 0; r < 8; ++r) acc[r] = 0.0f;

  const __bf16* Bb = xT + ((size_t)b * NN + n0) * NC;
  #pragma unroll
  for (int k = 0; k < NC; k += 32) {
    v16bf a = load_fragA(wq_bf + (size_t)o0 * NC + k, NC, lane);
    v16bf bb = load_fragB(Bb + k, NC, lane);
    acc = wmma_bf16f32(a, bb, acc);
  }

  // D layout: lane -> (N = lane&15, M = (lane>>4)*8 + r)
  const int n     = n0 + (lane & 15);
  const int obase = o0 + (lane >> 4) * 8;
  const int sec = obase >> 7;               // 0=Q 1=K 2=V
  const int h   = (obase >> 5) & 3;
  const int d0  = obase & 31;               // in {0,8,16,24}
  const size_t headNd = (((size_t)b * HEADS + h) * NN + n) * DHEAD;

  if (sec == 0) {
    v8bf pk;
    #pragma unroll
    for (int r = 0; r < 8; ++r) pk[r] = (__bf16)(acc[r] * QK_SCALE);
    *(v8bf*)(Qm + headNd + d0) = pk;
  } else if (sec == 1) {
    v8bf pk;
    #pragma unroll
    for (int r = 0; r < 8; ++r) pk[r] = (__bf16)acc[r];
    *(v8bf*)(Km + headNd + d0) = pk;
  } else {
    #pragma unroll
    for (int r = 0; r < 8; ++r)
      Vm[(((size_t)b * HEADS + h) * DHEAD + (d0 + r)) * NN + n] = (__bf16)acc[r];
  }
}

// ---------------------------------------------------------------- kernel 4
// Flash attention, one wave per (b, h, 32-query tile): two 16-query subtiles
// share every K/V fragment load (2x arithmetic intensity vs 16-query waves).
// Computes S^T = K_rows x Q so each softmax row lives in one lane + its
// lane^16 partner (single shuffle per reduction).  P^T staged via per-wave
// LDS tiles; out^T = V x P^T accumulated in four v8f tiles (2 subtiles x d32).
__global__ void flash_attn_k(const unsigned short* __restrict__ Qm_u,
                             const unsigned short* __restrict__ Km_u,
                             const unsigned short* __restrict__ Vm_u,
                             unsigned short* __restrict__ Om_u) {
  const __bf16* Qm = (const __bf16*)Qm_u;
  const __bf16* Km = (const __bf16*)Km_u;
  const __bf16* Vm = (const __bf16*)Vm_u;
  __bf16* Om = (__bf16*)Om_u;

  __shared__ __bf16 Pt[4][2][16][32];       // per-wave, per-subtile P^T, [i][j]

  const int wid  = threadIdx.x >> 5;
  const int lane = threadIdx.x & 31;
  const int wave = blockIdx.x * 4 + wid;
  const int it = wave & 127;                // 128 query tiles of 32
  const int t2 = wave >> 7;
  const int h = t2 & 3, b = t2 >> 2;
  const int i0 = it * 32;

  const __bf16* Qb = Qm + ((size_t)b * HEADS + h) * NN * DHEAD;   // [n][d]
  const __bf16* Kb = Km + ((size_t)b * HEADS + h) * NN * DHEAD;   // [n][d]
  const __bf16* Vb = Vm + ((size_t)b * HEADS + h) * DHEAD * NN;   // [d][n]

  // Q fragments: B operand (K=d 32 x N=i 16), constant over the whole loop.
  const v16bf qf0 = load_fragB(Qb + (size_t)i0 * DHEAD, DHEAD, lane);
  const v16bf qf1 = load_fragB(Qb + (size_t)(i0 + 16) * DHEAD, DHEAD, lane);

  v8f a00, a01, a10, a11, zf;               // a<subtile><d-half>
  #pragma unroll
  for (int r = 0; r < 8; ++r) { a00[r] = a01[r] = a10[r] = a11[r] = 0.0f; zf[r] = 0.0f; }
  float m0 = -3.0e38f, l0 = 0.0f;
  float m1 = -3.0e38f, l1 = 0.0f;

  const int ii = lane & 15;
  const int hb = (lane >> 4) * 8;

  for (int j0 = 0; j0 < NN; j0 += 32) {
    if (j0 + 32 < NN) {
      __builtin_prefetch(Kb + (size_t)(j0 + 32) * DHEAD, 0, 0);
      __builtin_prefetch(Vb + (j0 + 32), 0, 0);
    }
    // S^T tiles (M=j, N=i): A = K rows (shared by both subtiles), K-dim = d = 32.
    v16bf ka0 = load_fragA(Kb + (size_t)j0 * DHEAD, DHEAD, lane);
    v16bf ka1 = load_fragA(Kb + (size_t)(j0 + 16) * DHEAD, DHEAD, lane);
    v8f s00 = wmma_bf16f32(ka0, qf0, zf);
    v8f s01 = wmma_bf16f32(ka1, qf0, zf);
    v8f s10 = wmma_bf16f32(ka0, qf1, zf);
    v8f s11 = wmma_bf16f32(ka1, qf1, zf);

    // ---- online softmax, subtile 0 (rows i0+ii) ----
    {
      float mloc = s00[0];
      #pragma unroll
      for (int r = 1; r < 8; ++r) mloc = fmaxf(mloc, s00[r]);
      #pragma unroll
      for (int r = 0; r < 8; ++r) mloc = fmaxf(mloc, s01[r]);
      mloc = fmaxf(mloc, __shfl_xor(mloc, 16, 32));
      const float m_new = fmaxf(m0, mloc);
      const float resc = __expf(m0 - m_new);
      float lsum = 0.0f;
      v8bf p0, p1;
      #pragma unroll
      for (int r = 0; r < 8; ++r) { float p = __expf(s00[r] - m_new); lsum += p; p0[r] = (__bf16)p; }
      #pragma unroll
      for (int r = 0; r < 8; ++r) { float p = __expf(s01[r] - m_new); lsum += p; p1[r] = (__bf16)p; }
      lsum += __shfl_xor(lsum, 16, 32);
      l0 = l0 * resc + lsum;
      m0 = m_new;
      #pragma unroll
      for (int r = 0; r < 8; ++r) { a00[r] *= resc; a01[r] *= resc; }
      *(v8bf*)&Pt[wid][0][ii][hb]      = p0;
      *(v8bf*)&Pt[wid][0][ii][16 + hb] = p1;
    }
    // ---- online softmax, subtile 1 (rows i0+16+ii) ----
    {
      float mloc = s10[0];
      #pragma unroll
      for (int r = 1; r < 8; ++r) mloc = fmaxf(mloc, s10[r]);
      #pragma unroll
      for (int r = 0; r < 8; ++r) mloc = fmaxf(mloc, s11[r]);
      mloc = fmaxf(mloc, __shfl_xor(mloc, 16, 32));
      const float m_new = fmaxf(m1, mloc);
      const float resc = __expf(m1 - m_new);
      float lsum = 0.0f;
      v8bf p0, p1;
      #pragma unroll
      for (int r = 0; r < 8; ++r) { float p = __expf(s10[r] - m_new); lsum += p; p0[r] = (__bf16)p; }
      #pragma unroll
      for (int r = 0; r < 8; ++r) { float p = __expf(s11[r] - m_new); lsum += p; p1[r] = (__bf16)p; }
      lsum += __shfl_xor(lsum, 16, 32);
      l1 = l1 * resc + lsum;
      m1 = m_new;
      #pragma unroll
      for (int r = 0; r < 8; ++r) { a10[r] *= resc; a11[r] *= resc; }
      *(v8bf*)&Pt[wid][1][ii][hb]      = p0;
      *(v8bf*)&Pt[wid][1][ii][16 + hb] = p1;
    }
    asm volatile("s_wait_dscnt 0" ::: "memory");

    // out^T += V(d x j32) * P^T(j32 x i16); V fragments shared by both subtiles.
    v16bf pb0 = load_fragB(&Pt[wid][0][0][0], 32, lane);
    v16bf pb1 = load_fragB(&Pt[wid][1][0][0], 32, lane);
    v16bf va0 = load_fragA(Vb + j0, NN, lane);                    // d = 0..15
    v16bf va1 = load_fragA(Vb + (size_t)16 * NN + j0, NN, lane);  // d = 16..31
    a00 = wmma_bf16f32(va0, pb0, a00);
    a01 = wmma_bf16f32(va1, pb0, a01);
    a10 = wmma_bf16f32(va0, pb1, a10);
    a11 = wmma_bf16f32(va1, pb1, a11);
  }

  // normalize and write att-out as [b][n][inner] bf16 (B-operand friendly for out-proj)
  const float inv0 = 1.0f / l0, inv1 = 1.0f / l1;
  v8bf w00, w01, w10, w11;
  #pragma unroll
  for (int r = 0; r < 8; ++r) {
    w00[r] = (__bf16)(a00[r] * inv0);  w01[r] = (__bf16)(a01[r] * inv0);
    w10[r] = (__bf16)(a10[r] * inv1);  w11[r] = (__bf16)(a11[r] * inv1);
  }
  __bf16* op0 = Om + ((size_t)b * NN + (i0 + ii)) * INNER + h * DHEAD;
  *(v8bf*)(op0 + hb)      = w00;        // d = hb..hb+7
  *(v8bf*)(op0 + 16 + hb) = w01;        // d = 16+hb..
  __bf16* op1 = Om + ((size_t)b * NN + (i0 + 16 + ii)) * INNER + h * DHEAD;
  *(v8bf*)(op1 + hb)      = w10;
  *(v8bf*)(op1 + 16 + hb) = w11;
}

// ---------------------------------------------------------------- kernel 5
// Output projection: y[b][o][n] = sum_c w_out[o][c]*att[b][n][c] + b_out[o].
__global__ void out_proj_k(const unsigned short* __restrict__ wo_bf_u,
                           const unsigned short* __restrict__ Om_u,
                           const float* __restrict__ b_out,
                           float* __restrict__ y) {
  const __bf16* wo_bf = (const __bf16*)wo_bf_u;
  const __bf16* Om    = (const __bf16*)Om_u;

  const int lane = threadIdx.x & 31;
  const int wave = (blockIdx.x * blockDim.x + threadIdx.x) >> 5;
  const int nt = wave & 255;
  const int t2 = wave >> 8;
  const int ot = t2 & 15;
  const int b  = t2 >> 4;
  const int o0 = ot * 16, n0 = nt * 16;

  v8f acc;
  #pragma unroll
  for (int r = 0; r < 8; ++r) acc[r] = 0.0f;

  const __bf16* Bb = Om + ((size_t)b * NN + n0) * INNER;
  #pragma unroll
  for (int k = 0; k < INNER; k += 32) {
    v16bf a  = load_fragA(wo_bf + (size_t)o0 * INNER + k, INNER, lane);
    v16bf bb = load_fragB(Bb + k, INNER, lane);
    acc = wmma_bf16f32(a, bb, acc);
  }

  const int n  = n0 + (lane & 15);
  const int ob = o0 + (lane >> 4) * 8;
  float* yp = y + ((size_t)b * NC + ob) * NN + n;
  #pragma unroll
  for (int r = 0; r < 8; ++r) yp[(size_t)r * NN] = acc[r] + b_out[ob + r];
}

// ---------------------------------------------------------------- launch
extern "C" void kernel_launch(void* const* d_in, const int* in_sizes, int n_in,
                              void* d_out, int out_size, void* d_ws, size_t ws_size,
                              hipStream_t stream) {
  const float* x     = (const float*)d_in[0];
  const float* w_qkv = (const float*)d_in[1];
  const float* w_out = (const float*)d_in[2];
  const float* b_out = (const float*)d_in[3];
  float* y = (float*)d_out;

  char* ws = (char*)d_ws;
  size_t off = 0;
  unsigned short* wq_bf = (unsigned short*)(ws + off); off += (size_t)OQKV * NC * 2;     // 192 KB
  unsigned short* wo_bf = (unsigned short*)(ws + off); off += (size_t)NC * INNER * 2;    // 64 KB
  unsigned short* xT    = (unsigned short*)(ws + off); off += (size_t)NB * NN * NC * 2;  // 8 MB
  unsigned short* Qm    = (unsigned short*)(ws + off); off += (size_t)NB * HEADS * NN * DHEAD * 2; // 4 MB
  unsigned short* Km    = (unsigned short*)(ws + off); off += (size_t)NB * HEADS * NN * DHEAD * 2; // 4 MB
  unsigned short* Vm    = (unsigned short*)(ws + off); off += (size_t)NB * HEADS * NN * DHEAD * 2; // 4 MB
  unsigned short* Om    = (unsigned short*)(ws + off); off += (size_t)NB * NN * INNER * 2;         // 4 MB

  cvt_weights_k<<<(OQKV * NC + 255) / 256, 256, 0, stream>>>(w_qkv, w_out, wq_bf, wo_bf);

  transpose_x_k<<<dim3(NN / 32, NC / 32, NB), dim3(32, 8, 1), 0, stream>>>(x, xT);

  // NB * 24 o-tiles * 256 n-tiles waves, 4 waves/block
  qkv_proj_k<<<(NB * 24 * 256) / 4, 128, 0, stream>>>(wq_bf, xT, Qm, Km, Vm);

  // NB * HEADS * 128 query tiles (of 32) waves, 4 waves/block
  flash_attn_k<<<(NB * HEADS * 128) / 4, 128, 0, stream>>>(Qm, Km, Vm, Om);

  // NB * 16 o-tiles * 256 n-tiles waves, 4 waves/block
  out_proj_k<<<(NB * 16 * 256) / 4, 128, 0, stream>>>(wo_bf, Om, b_out, y);
}
